// LTCCell_44178033607122
// MI455X (gfx1250) — matrix operand order repaired and checked by reading, not compiled, not run on hardware
//
#include <hip/hip_runtime.h>
#include <hip/hip_bf16.h>
#include <math.h>

// ---------------------------------------------------------------------------
// LTC cell, MI455X (gfx1250): bf16 WMMA GEMMs (fp32 accumulate), async
// global->LDS double-buffered pipeline, fused fast-transcendental epilogues.
// B=8192, I=512, H=1024, K=I+H=1536.
// ---------------------------------------------------------------------------

typedef __bf16 v16bf __attribute__((ext_vector_type(16)));
typedef float  v8f   __attribute__((ext_vector_type(8)));
typedef int    v4i_vs __attribute__((vector_size(16)));   // matches builtin param

#define B_DIM   8192
#define I_DIM   512
#define H_DIM   1024
#define K_DIM   1536
#define KSTEPS  48            // K_DIM / 32
#define ROWU4   192           // K_DIM halfs / 8 per uint4
#define LDSROW  5             // uint4 per LDS row: 32 halfs + 8 pad halfs
#define EPS     1e-6f

#if defined(__has_builtin)
#if __has_builtin(__builtin_amdgcn_global_load_async_to_lds_b128) && \
    __has_builtin(__builtin_amdgcn_s_wait_asynccnt)
#define HAVE_ASYNC_LDS 1
#endif
#endif

union FragBF { uint4 u[2]; v16bf v; };

__device__ __forceinline__ unsigned short f2bf(float f) {
  unsigned int u = __float_as_uint(f);
  u += 0x7fffu + ((u >> 16) & 1u);          // round-to-nearest-even
  return (unsigned short)(u >> 16);
}
__device__ __forceinline__ unsigned int f2bf2(float lo, float hi) {
  return (unsigned int)f2bf(lo) | ((unsigned int)f2bf(hi) << 16);
}

// fast transcendentals (hardware v_exp_f32 / v_log_f32 / v_rcp_f32 / v_tanh)
__device__ __forceinline__ float fast_sigmoid(float x) {
  return __builtin_amdgcn_rcpf(1.0f + __expf(-x));
}
__device__ __forceinline__ float fast_softplus(float x) {
  return fmaxf(x, 0.0f) + __logf(1.0f + __expf(-fabsf(x)));
}
__device__ __forceinline__ float fast_tanh(float x) {
#if defined(__has_builtin) && __has_builtin(__builtin_amdgcn_tanhf)
  return __builtin_amdgcn_tanhf(x);
#elif defined(__has_builtin) && __has_builtin(__builtin_amdgcn_tanh_f32)
  return __builtin_amdgcn_tanh_f32(x);
#else
  float e = __expf(2.0f * x);
  return 1.0f - 2.0f * __builtin_amdgcn_rcpf(e + 1.0f);
#endif
}

#if HAVE_ASYNC_LDS
__device__ __forceinline__ void async_cp16(const uint4* g, uint4* l) {
  __builtin_amdgcn_global_load_async_to_lds_b128(
      (__attribute__((address_space(1))) v4i_vs*)g,   // C-style cast drops const
      (__attribute__((address_space(3))) v4i_vs*)l,
      0, 0);
}
#endif

// ---------------------------------------------------------------------------
// Pack [x | hidden] -> combined (bf16) and x -> cand[:, :I] (bf16).
// 8 elements per thread, uint4 (8xbf16) stores.
// ---------------------------------------------------------------------------
__global__ __launch_bounds__(256) void ltc_pack_combined(
    const float* __restrict__ x, const float* __restrict__ hidden,
    unsigned short* __restrict__ combined, unsigned short* __restrict__ cand) {
  int idx = blockIdx.x * 256 + threadIdx.x;       // chunk of 8 halfs
  int row = idx / (K_DIM / 8);
  int cc  = idx - row * (K_DIM / 8);
  int col = cc * 8;
  float4 f0, f1;
  if (col < I_DIM) {
    const float4* p = (const float4*)(x + (size_t)row * I_DIM + col);
    f0 = p[0]; f1 = p[1];
  } else {
    const float4* p = (const float4*)(hidden + (size_t)row * H_DIM + (col - I_DIM));
    f0 = p[0]; f1 = p[1];
  }
  uint4 u;
  u.x = f2bf2(f0.x, f0.y); u.y = f2bf2(f0.z, f0.w);
  u.z = f2bf2(f1.x, f1.y); u.w = f2bf2(f1.z, f1.w);
  ((uint4*)combined)[idx] = u;
  if (col < I_DIM) ((uint4*)cand)[idx] = u;
}

// fp32 -> bf16 (weights), 8 elems/thread
__global__ __launch_bounds__(256) void ltc_cvt_bf16(
    const float* __restrict__ src, unsigned short* __restrict__ dst) {
  int idx = blockIdx.x * 256 + threadIdx.x;
  const float4* p = (const float4*)(src + (size_t)idx * 8);
  float4 f0 = p[0], f1 = p[1];
  uint4 u;
  u.x = f2bf2(f0.x, f0.y); u.y = f2bf2(f0.z, f0.w);
  u.z = f2bf2(f1.x, f1.y); u.w = f2bf2(f1.z, f1.w);
  ((uint4*)dst)[idx] = u;
}

// ---------------------------------------------------------------------------
// Kernel 1: tau = softplus(combined @ Wtau^T + b_tau) + eps   (store fp32)
//           r   = sigmoid (combined @ Wr^T   + b_r)
//           cand[:, I:] = bf16(r * hidden)
// Block tile 128x128, 8 waves, each wave 64x32 (4x2 WMMA tiles) per gate.
// Async global->LDS, double-buffered, one barrier per K-step.
// ---------------------------------------------------------------------------
__global__ __launch_bounds__(256) void ltc_gemm_taur(
    const unsigned short* __restrict__ combined,
    const unsigned short* __restrict__ Wtau,
    const unsigned short* __restrict__ Wr,
    const float* __restrict__ b_tau, const float* __restrict__ b_r,
    const float* __restrict__ hidden,
    float* __restrict__ tau_out, unsigned short* __restrict__ cand) {
#if HAVE_ASYNC_LDS
  __shared__ uint4 sA [2][128 * LDSROW];
  __shared__ uint4 sBt[2][128 * LDSROW];
  __shared__ uint4 sBr[2][128 * LDSROW];
#else
  __shared__ uint4 sA [1][128 * LDSROW];
  __shared__ uint4 sBt[1][128 * LDSROW];
  __shared__ uint4 sBr[1][128 * LDSROW];
#endif

  const int tid  = threadIdx.x;
  const int lane = tid & 31;
  const int w    = tid >> 5;
  const int wm   = w >> 2;                // 0..1 : 64 rows each
  const int wn   = w & 3;                 // 0..3 : 32 cols each
  const int lm   = lane & 15;
  const int kh   = lane >> 4;

  const int row0 = blockIdx.y * 128;
  const int col0 = blockIdx.x * 128;

  const uint4* gA  = (const uint4*)combined;
  const uint4* gWt = (const uint4*)Wtau;
  const uint4* gWr = (const uint4*)Wr;

  v8f accT[4][2], accR[4][2];
#pragma unroll
  for (int i = 0; i < 4; ++i)
#pragma unroll
    for (int j = 0; j < 2; ++j) { accT[i][j] = (v8f)0.0f; accR[i][j] = (v8f)0.0f; }

#if HAVE_ASYNC_LDS
  // ---- async double-buffered pipeline -------------------------------------
  #define TAUR_ISSUE(KT, BUF)                                                  \
    {                                                                          \
      _Pragma("unroll")                                                        \
      for (int it = 0; it < 2; ++it) {                                         \
        int ch = tid + it * 256;                                               \
        int m = ch >> 2, c = ch & 3;                                           \
        async_cp16(gA  + (size_t)(row0 + m) * ROWU4 + (KT) * 4 + c,            \
                   &sA [BUF][m * LDSROW + c]);                                 \
        async_cp16(gWt + (size_t)(col0 + m) * ROWU4 + (KT) * 4 + c,            \
                   &sBt[BUF][m * LDSROW + c]);                                 \
        async_cp16(gWr + (size_t)(col0 + m) * ROWU4 + (KT) * 4 + c,            \
                   &sBr[BUF][m * LDSROW + c]);                                 \
      }                                                                        \
    }
  TAUR_ISSUE(0, 0);
  __builtin_amdgcn_s_wait_asynccnt(0);
  __syncthreads();
#endif

  for (int kt = 0; kt < KSTEPS; ++kt) {
#if HAVE_ASYNC_LDS
    const int cb = kt & 1;
    if (kt + 1 < KSTEPS) TAUR_ISSUE(kt + 1, cb ^ 1);   // prefetch next tile
#else
    const int cb = 0;
    __syncthreads();
#pragma unroll
    for (int it = 0; it < 2; ++it) {
      int ch = tid + it * 256;
      int m = ch >> 2, c = ch & 3;
      sA [0][m * LDSROW + c] = gA [(size_t)(row0 + m) * ROWU4 + kt * 4 + c];
      sBt[0][m * LDSROW + c] = gWt[(size_t)(col0 + m) * ROWU4 + kt * 4 + c];
      sBr[0][m * LDSROW + c] = gWr[(size_t)(col0 + m) * ROWU4 + kt * 4 + c];
    }
    __syncthreads();
#endif

    FragBF a[4], bt[2], br[2];
#pragma unroll
    for (int i = 0; i < 4; ++i) {         // A: row lm, K halves by lane group
      int rb = (wm * 64 + i * 16 + lm) * LDSROW;
      a[i].u[0] = sA[cb][rb + kh * 2];
      a[i].u[1] = sA[cb][rb + kh * 2 + 1];
    }
#pragma unroll
    for (int j = 0; j < 2; ++j) {         // B: W-row per lane, contiguous 16 K
      int rb = (wn * 32 + j * 16 + lm) * LDSROW;
      bt[j].u[0] = sBt[cb][rb + kh * 2];
      bt[j].u[1] = sBt[cb][rb + kh * 2 + 1];
      br[j].u[0] = sBr[cb][rb + kh * 2];
      br[j].u[1] = sBr[cb][rb + kh * 2 + 1];
    }
#pragma unroll
    for (int i = 0; i < 4; ++i)
#pragma unroll
      for (int j = 0; j < 2; ++j) {
        accT[i][j] = __builtin_amdgcn_wmma_f32_16x16x32_bf16(
            false, a[i].v, false, bt[j].v, (short)0, accT[i][j], false, false);
        accR[i][j] = __builtin_amdgcn_wmma_f32_16x16x32_bf16(
            false, a[i].v, false, br[j].v, (short)0, accR[i][j], false, false);
      }

#if HAVE_ASYNC_LDS
    if (kt + 1 < KSTEPS) {
      __builtin_amdgcn_s_wait_asynccnt(0);   // my prefetch landed
      __syncthreads();                       // everyone's landed / done reading
    }
#endif
  }

  // Epilogue: C/D layout -> row = base + r + 8*(lane>=16), col = base + lane&15
#pragma unroll
  for (int i = 0; i < 4; ++i)
#pragma unroll
    for (int j = 0; j < 2; ++j)
#pragma unroll
      for (int r = 0; r < 8; ++r) {
        int grow = row0 + wm * 64 + i * 16 + r + kh * 8;
        int gcol = col0 + wn * 32 + j * 16 + lm;
        size_t o = (size_t)grow * H_DIM + gcol;
        float vt = accT[i][j][r] + b_tau[gcol];
        tau_out[o] = fast_softplus(vt) + EPS;
        float vr = accR[i][j][r] + b_r[gcol];
        cand[(size_t)grow * K_DIM + I_DIM + gcol] =
            f2bf(fast_sigmoid(vr) * hidden[o]);
      }
}

// ---------------------------------------------------------------------------
// Kernel 2: h_tilde = tanh(cand @ Wh^T + b_h); out = h + (h_tilde - h)/tau
// ---------------------------------------------------------------------------
__global__ __launch_bounds__(256) void ltc_gemm_h(
    const unsigned short* __restrict__ cand,
    const unsigned short* __restrict__ Wh,
    const float* __restrict__ b_h,
    const float* __restrict__ hidden, const float* __restrict__ tau,
    float* __restrict__ out) {
#if HAVE_ASYNC_LDS
  __shared__ uint4 sA[2][128 * LDSROW];
  __shared__ uint4 sB[2][128 * LDSROW];
#else
  __shared__ uint4 sA[1][128 * LDSROW];
  __shared__ uint4 sB[1][128 * LDSROW];
#endif

  const int tid  = threadIdx.x;
  const int lane = tid & 31;
  const int w    = tid >> 5;
  const int wm   = w >> 2;
  const int wn   = w & 3;
  const int lm   = lane & 15;
  const int kh   = lane >> 4;

  const int row0 = blockIdx.y * 128;
  const int col0 = blockIdx.x * 128;

  const uint4* gA = (const uint4*)cand;
  const uint4* gW = (const uint4*)Wh;

  v8f acc[4][2];
#pragma unroll
  for (int i = 0; i < 4; ++i)
#pragma unroll
    for (int j = 0; j < 2; ++j) acc[i][j] = (v8f)0.0f;

#if HAVE_ASYNC_LDS
  #define H_ISSUE(KT, BUF)                                                     \
    {                                                                          \
      _Pragma("unroll")                                                        \
      for (int it = 0; it < 2; ++it) {                                         \
        int ch = tid + it * 256;                                               \
        int m = ch >> 2, c = ch & 3;                                           \
        async_cp16(gA + (size_t)(row0 + m) * ROWU4 + (KT) * 4 + c,             \
                   &sA[BUF][m * LDSROW + c]);                                  \
        async_cp16(gW + (size_t)(col0 + m) * ROWU4 + (KT) * 4 + c,             \
                   &sB[BUF][m * LDSROW + c]);                                  \
      }                                                                        \
    }
  H_ISSUE(0, 0);
  __builtin_amdgcn_s_wait_asynccnt(0);
  __syncthreads();
#endif

  for (int kt = 0; kt < KSTEPS; ++kt) {
#if HAVE_ASYNC_LDS
    const int cb = kt & 1;
    if (kt + 1 < KSTEPS) H_ISSUE(kt + 1, cb ^ 1);
#else
    const int cb = 0;
    __syncthreads();
#pragma unroll
    for (int it = 0; it < 2; ++it) {
      int ch = tid + it * 256;
      int m = ch >> 2, c = ch & 3;
      sA[0][m * LDSROW + c] = gA[(size_t)(row0 + m) * ROWU4 + kt * 4 + c];
      sB[0][m * LDSROW + c] = gW[(size_t)(col0 + m) * ROWU4 + kt * 4 + c];
    }
    __syncthreads();
#endif

    FragBF a[4], b[2];
#pragma unroll
    for (int i = 0; i < 4; ++i) {
      int rb = (wm * 64 + i * 16 + lm) * LDSROW;
      a[i].u[0] = sA[cb][rb + kh * 2];
      a[i].u[1] = sA[cb][rb + kh * 2 + 1];
    }
#pragma unroll
    for (int j = 0; j < 2; ++j) {
      int rb = (wn * 32 + j * 16 + lm) * LDSROW;
      b[j].u[0] = sB[cb][rb + kh * 2];
      b[j].u[1] = sB[cb][rb + kh * 2 + 1];
    }
#pragma unroll
    for (int i = 0; i < 4; ++i)
#pragma unroll
      for (int j = 0; j < 2; ++j)
        acc[i][j] = __builtin_amdgcn_wmma_f32_16x16x32_bf16(
            false, a[i].v, false, b[j].v, (short)0, acc[i][j], false, false);

#if HAVE_ASYNC_LDS
    if (kt + 1 < KSTEPS) {
      __builtin_amdgcn_s_wait_asynccnt(0);
      __syncthreads();
    }
#endif
  }

#pragma unroll
  for (int i = 0; i < 4; ++i)
#pragma unroll
    for (int j = 0; j < 2; ++j)
#pragma unroll
      for (int r = 0; r < 8; ++r) {
        int grow = row0 + wm * 64 + i * 16 + r + kh * 8;
        int gcol = col0 + wn * 32 + j * 16 + lm;
        size_t o = (size_t)grow * H_DIM + gcol;
        float ht = fast_tanh(acc[i][j][r] + b_h[gcol]);
        float h  = hidden[o];
        out[o] = h + (ht - h) * __builtin_amdgcn_rcpf(tau[o]);
      }
}

// ---------------------------------------------------------------------------
extern "C" void kernel_launch(void* const* d_in, const int* in_sizes, int n_in,
                              void* d_out, int out_size, void* d_ws, size_t ws_size,
                              hipStream_t stream) {
  const float* x      = (const float*)d_in[0];
  const float* hidden = (const float*)d_in[1];
  const float* W_tau  = (const float*)d_in[2];
  const float* b_tau  = (const float*)d_in[3];
  const float* W_r    = (const float*)d_in[4];
  const float* b_r    = (const float*)d_in[5];
  const float* W_h    = (const float*)d_in[6];
  const float* b_h    = (const float*)d_in[7];
  float* out = (float*)d_out;

  // Workspace layout (bytes)
  char* ws = (char*)d_ws;
  unsigned short* combined = (unsigned short*)ws;                         // B*K
  unsigned short* cand     = (unsigned short*)(ws + (size_t)B_DIM * K_DIM * 2);
  unsigned short* wtau_bf  = (unsigned short*)(ws + (size_t)B_DIM * K_DIM * 4);
  unsigned short* wr_bf    = wtau_bf + (size_t)H_DIM * K_DIM;
  unsigned short* wh_bf    = wr_bf   + (size_t)H_DIM * K_DIM;
  float*          tau_buf  = (float*)(wh_bf + (size_t)H_DIM * K_DIM);    // B*H

  // 1) bf16 packing / conversion (bandwidth-bound, ~µs)
  {
    int n8 = B_DIM * K_DIM / 8;
    ltc_pack_combined<<<n8 / 256, 256, 0, stream>>>(x, hidden, combined, cand);
    int nw8 = H_DIM * K_DIM / 8;
    ltc_cvt_bf16<<<nw8 / 256, 256, 0, stream>>>(W_tau, wtau_bf);
    ltc_cvt_bf16<<<nw8 / 256, 256, 0, stream>>>(W_r,   wr_bf);
    ltc_cvt_bf16<<<nw8 / 256, 256, 0, stream>>>(W_h,   wh_bf);
  }

  // 2) fused tau + r GEMMs, write tau and gated (cand[:, I:])
  {
    dim3 grid(H_DIM / 128, B_DIM / 128);   // 8 x 64 blocks
    ltc_gemm_taur<<<grid, 256, 0, stream>>>(combined, wtau_bf, wr_bf,
                                            b_tau, b_r, hidden, tau_buf, cand);
  }

  // 3) candidate GEMM + final LTC update
  {
    dim3 grid(H_DIM / 128, B_DIM / 128);
    ltc_gemm_h<<<grid, 256, 0, stream>>>(cand, wh_bf, b_h, hidden, tau_buf, out);
  }
}
